// Encoder_83330955477074
// MI455X (gfx1250) — matrix operand order repaired and checked by reading, not compiled, not run on hardware
//
#include <hip/hip_runtime.h>
#include <hip/hip_bf16.h>
#include <stdint.h>

// Problem constants (match reference)
#define B_  128
#define S_  128
#define IN_ 2
#define H_  512
#define SS_ 100
#define L_  128
#define H3_ (3 * H_)

typedef __attribute__((ext_vector_type(16))) _Float16 v16h;
typedef __attribute__((ext_vector_type(8)))  _Float16 v8h;
typedef __attribute__((ext_vector_type(8)))  float    v8f;

// Async copy: 4 bytes global -> LDS, tracked by ASYNCcnt (CDNA5 §15.18 op 96).
// ldsoff = low 32 bits of flat LDS pointer (aperture maps addr[31:0] -> LDS).
__device__ __forceinline__ void async_copy_b32(void* lds_dst, const void* gsrc) {
  uint32_t dst = (uint32_t)(uintptr_t)lds_dst;
  uint64_t src = (uint64_t)(uintptr_t)gsrc;
  asm volatile("global_load_async_to_lds_b32 %0, %1, off"
               :: "v"(dst), "v"(src) : "memory");
}
__device__ __forceinline__ void wait_async0() {
  asm volatile("s_wait_asynccnt 0x0" ::: "memory");
}

// ---------------------------------------------------------------------------
// f32 -> f16 conversion (weights + instance_hidden), one-time
// ---------------------------------------------------------------------------
__global__ void cvt_f32_to_f16(const float* __restrict__ src,
                               _Float16* __restrict__ dst, int n) {
  int t = blockIdx.x * blockDim.x + threadIdx.x;
  if (t < n) dst[t] = (_Float16)src[t];
}

// ---------------------------------------------------------------------------
// ref_emb[b,l,:] = instance[b, solution[b,l], :] @ Wr^T + br   (IN=2, to f16)
// ---------------------------------------------------------------------------
__global__ __launch_bounds__(H_)
void embed_gather(const float* __restrict__ inst, const int* __restrict__ sol,
                  const float* __restrict__ Wr, const float* __restrict__ br,
                  _Float16* __restrict__ ref_h) {
  int bl = blockIdx.x;            // b*L + l
  int b  = bl >> 7;
  int l  = bl & (L_ - 1);
  int idx = sol[b * S_ + l];
  float x0 = inst[((size_t)b * S_ + idx) * IN_ + 0];
  float x1 = inst[((size_t)b * S_ + idx) * IN_ + 1];
  int h = threadIdx.x;
  float r = x0 * Wr[h * IN_ + 0] + x1 * Wr[h * IN_ + 1] + br[h];
  ref_h[(size_t)bl * H_ + h] = (_Float16)r;
}

// ---------------------------------------------------------------------------
// zero-initialize hidden states (f32 + f16 mirrors)
// ---------------------------------------------------------------------------
__global__ void zero_states(float* h1f, float* h2f,
                            _Float16* h1h, _Float16* h2h) {
  int t = blockIdx.x * blockDim.x + threadIdx.x;
  h1f[t] = 0.f; h2f[t] = 0.f;
  h1h[t] = (_Float16)0.f; h2h[t] = (_Float16)0.f;
}

// ---------------------------------------------------------------------------
// WMMA GEMM: C(M,Ntot) = A(M,K)_f16 @ W(Ntot,K)_f16^T + bias
//   grid = (Ntot/256, M/16), block = 256 threads = 8 waves
//   wave tile: 16(M) x 32(N) -> two 16x16 f32 accumulators sharing one A frag
//   A slab (16x32 f16) double-buffered in LDS via GLOBAL_LOAD_ASYNC_TO_LDS
// Fragment layouts used here (wave32):
//   A: lane L -> M = L&15; 16 contiguous K starting at (L<16 ? 0 : 16)
//   B: lane L -> N = L&15; 16 contiguous K starting at (L<16 ? 0 : 16)
//   C: lane L -> N = L&15; VGPR j -> M = j + (L<16 ? 0 : 8)
// ---------------------------------------------------------------------------
__global__ __launch_bounds__(256)
void wmma_gemm_f16(const _Float16* __restrict__ A, int lda,
                   const _Float16* __restrict__ W,       // (Ntot, K) row-major
                   const float* __restrict__ bias,       // may be null
                   float* __restrict__ C, int ldc, int K) {
  __shared__ __align__(16) _Float16 As[2][16][40];   // padded rows (80B stride)

  const int tid  = threadIdx.x;
  const int lane = tid & 31;
  const int wave = tid >> 5;
  const int m0   = blockIdx.y << 4;
  const int n0   = (blockIdx.x << 8) + (wave << 5);  // 32 N-cols per wave
  const int mloc = lane & 15;
  const int half = lane >> 4;       // 0 or 1
  const int abase = half << 3;      // unused; A also uses kbase (0 / 16)
  const int kbase = half << 4;      // B K-offset: 0 / 16

  const int arow = tid >> 4;        // 0..15 (cooperative A stage)
  const int acol = (tid & 15) << 1; // 0..30, step 2

  const _Float16* agbase = A + (size_t)(m0 + arow) * lda + acol;

  // prologue: stage slab 0 asynchronously
  async_copy_b32(&As[0][arow][acol], agbase);
  wait_async0();
  __syncthreads();

  v8f acc0 = {}, acc1 = {};
  const int steps = K >> 5;
  for (int i = 0; i < steps; ++i) {
    const int cur = i & 1;
    // prefetch next 16x32 A slab into the other buffer while we compute
    if (i + 1 < steps)
      async_copy_b32(&As[cur ^ 1][arow][acol], agbase + ((i + 1) << 5));

    v16h afrag;
    {
      v8h lo  = *(const v8h*)&As[cur][mloc][kbase];
      v8h hi8 = *(const v8h*)&As[cur][mloc][kbase + 8];
      #pragma unroll
      for (int e = 0; e < 8; ++e) { afrag[e] = lo[e]; afrag[e + 8] = hi8[e]; }
    }
    const _Float16* wrow = W + (size_t)(n0 + mloc) * K + (i << 5) + kbase;
    v16h b0 = *(const v16h*)wrow;                      // cols n0 .. n0+15
    v16h b1 = *(const v16h*)(wrow + (size_t)16 * K);   // cols n0+16 .. n0+31

    acc0 = __builtin_amdgcn_wmma_f32_16x16x32_f16(
        false, afrag, false, b0, (short)0, acc0, false, false);
    acc1 = __builtin_amdgcn_wmma_f32_16x16x32_f16(
        false, afrag, false, b1, (short)0, acc1, false, false);

    // next slab must be resident and everyone done reading before flip
    wait_async0();
    __syncthreads();
  }

  const int n = n0 + mloc;
  const float bv0 = bias ? bias[n] : 0.0f;
  const float bv1 = bias ? bias[n + 16] : 0.0f;
  #pragma unroll
  for (int j = 0; j < 8; ++j) {
    int m = m0 + j + (half << 3);
    C[(size_t)m * ldc + n]      = acc0[j] + bv0;
    C[(size_t)m * ldc + n + 16] = acc1[j] + bv1;
  }
}

// ---------------------------------------------------------------------------
// GRU gate fusion: h = (1-z)*n + z*h, gates from gi(B,3H)+gh(B,3H)
// ---------------------------------------------------------------------------
__global__ __launch_bounds__(256)
void gru_update(const float* __restrict__ gi, const float* __restrict__ gh,
                float* __restrict__ hf, _Float16* __restrict__ hh) {
  int t = blockIdx.x * blockDim.x + threadIdx.x;   // B*H
  int b = t >> 9;
  int h = t & (H_ - 1);
  size_t g = (size_t)b * H3_;
  float ir = gi[g + h], iz = gi[g + H_ + h], in = gi[g + 2 * H_ + h];
  float hr = gh[g + h], hz = gh[g + H_ + h], hn = gh[g + 2 * H_ + h];
  float r = 1.f / (1.f + __expf(-(ir + hr)));
  float z = 1.f / (1.f + __expf(-(iz + hz)));
  float n = tanhf(in + r * hn);
  float hv = (1.f - z) * n + z * hf[t];
  hf[t] = hv;
  hh[t] = (_Float16)hv;
}

// ---------------------------------------------------------------------------
// Attention step: scores -> softmax -> context; also builds GRU2 input
// one block per batch b, 512 threads (16 waves)
// ---------------------------------------------------------------------------
__global__ __launch_bounds__(H_)
void attn_step(const float* __restrict__ inst_proj,  // (B,S,H) f32
               const float* __restrict__ uah,        // (B,H)  h1@Ua^T
               const float* __restrict__ vvec,       // (H)
               const float* __restrict__ ih,         // (B,S,H) f32
               const _Float16* __restrict__ ref_h,   // (B,L,H) f16
               _Float16* __restrict__ cat2,          // (B,2H) f16 GRU2 input
               int l) {
  __shared__ float uahs[H_];
  __shared__ float vs[H_];
  __shared__ float sc[S_];

  int b = blockIdx.x;
  int tid = threadIdx.x, lane = tid & 31, wave = tid >> 5;

  uahs[tid] = uah[(size_t)b * H_ + tid];
  vs[tid]   = vvec[tid];
  __syncthreads();

  // scores[s] = sum_h v[h] * tanh(inst_proj[b,s,h] + uah[b,h])
  for (int s = wave; s < S_; s += 16) {
    const float* row = inst_proj + ((size_t)b * S_ + s) * H_;
    float p = 0.f;
    for (int h = lane; h < H_; h += 32)
      p += vs[h] * tanhf(row[h] + uahs[h]);
    #pragma unroll
    for (int m = 16; m; m >>= 1) p += __shfl_xor(p, m, 32);
    if (lane == 0) sc[s] = p;
  }
  __syncthreads();

  // softmax over S=128 (wave 0, 4 elements per lane)
  if (wave == 0) {
    float v0 = sc[lane], v1 = sc[lane + 32], v2 = sc[lane + 64], v3 = sc[lane + 96];
    float mx = fmaxf(fmaxf(v0, v1), fmaxf(v2, v3));
    #pragma unroll
    for (int m = 16; m; m >>= 1) mx = fmaxf(mx, __shfl_xor(mx, m, 32));
    v0 = __expf(v0 - mx); v1 = __expf(v1 - mx);
    v2 = __expf(v2 - mx); v3 = __expf(v3 - mx);
    float sm = v0 + v1 + v2 + v3;
    #pragma unroll
    for (int m = 16; m; m >>= 1) sm += __shfl_xor(sm, m, 32);
    float inv = 1.f / sm;
    sc[lane] = v0 * inv; sc[lane + 32] = v1 * inv;
    sc[lane + 64] = v2 * inv; sc[lane + 96] = v3 * inv;
  }
  __syncthreads();

  // context[h] = sum_s attn[s] * ih[b,s,h]; GRU2 input = [x2ref | context]
  int h = tid;
  const float* ihb = ih + (size_t)b * S_ * H_ + h;
  float acc = 0.f;
  for (int s = 0; s < S_; ++s) acc += sc[s] * ihb[(size_t)s * H_];
  cat2[(size_t)b * 2 * H_ + H_ + h] = (_Float16)acc;
  cat2[(size_t)b * 2 * H_ + h] = ref_h[((size_t)b * L_ + (l + 1)) * H_ + h];
}

// ---------------------------------------------------------------------------
// Output head: mu = h2@W1^T+b1, lv = h2@W2^T+b2, z = mu + eps*exp(0.5*lv)
// out = [z | mu | lv], each (B,SS)
// ---------------------------------------------------------------------------
__global__ __launch_bounds__(128)
void final_head(const float* __restrict__ h2,
                const float* __restrict__ W1, const float* __restrict__ b1,
                const float* __restrict__ W2, const float* __restrict__ b2,
                const float* __restrict__ eps, float* __restrict__ out) {
  int b = blockIdx.x;
  int ss = threadIdx.x;
  if (ss >= SS_) return;
  const float* hrow = h2 + (size_t)b * H_;
  const float* w1 = W1 + (size_t)ss * H_;
  const float* w2 = W2 + (size_t)ss * H_;
  float mu = b1[ss], lv = b2[ss];
  for (int h = 0; h < H_; ++h) {
    float hv = hrow[h];
    mu += hv * w1[h];
    lv += hv * w2[h];
  }
  float z = mu + eps[(size_t)b * SS_ + ss] * __expf(0.5f * lv);
  out[(size_t)b * SS_ + ss] = z;
  out[(size_t)(B_ + b) * SS_ + ss] = mu;
  out[(size_t)(2 * B_ + b) * SS_ + ss] = lv;
}

// ---------------------------------------------------------------------------
// host driver
// ---------------------------------------------------------------------------
extern "C" void kernel_launch(void* const* d_in, const int* in_sizes, int n_in,
                              void* d_out, int out_size, void* d_ws, size_t ws_size,
                              hipStream_t stream) {
  (void)in_sizes; (void)n_in; (void)out_size; (void)ws_size;

  const float* instance = (const float*)d_in[0];
  const int*   solution = (const int*)  d_in[1];
  const float* ih       = (const float*)d_in[2];
  const float* eps      = (const float*)d_in[3];
  const float* Wr  = (const float*)d_in[4];
  const float* br  = (const float*)d_in[5];
  const float* Wi1 = (const float*)d_in[6];
  const float* Wh1 = (const float*)d_in[7];
  const float* bi1 = (const float*)d_in[8];
  const float* bh1 = (const float*)d_in[9];
  const float* Wa  = (const float*)d_in[10];
  const float* Ua  = (const float*)d_in[11];
  const float* vv  = (const float*)d_in[12];
  const float* Wi2 = (const float*)d_in[13];
  const float* Wh2 = (const float*)d_in[14];
  const float* bi2 = (const float*)d_in[15];
  const float* bh2 = (const float*)d_in[16];
  const float* W1  = (const float*)d_in[17];
  const float* b1  = (const float*)d_in[18];
  const float* W2  = (const float*)d_in[19];
  const float* b2  = (const float*)d_in[20];
  float* out = (float*)d_out;

  // carve workspace
  char* p = (char*)d_ws;
  auto alloc = [&](size_t bytes) -> void* {
    void* r = (void*)p;
    p += (bytes + 255) & ~(size_t)255;
    return r;
  };
  _Float16* Wi1h = (_Float16*)alloc((size_t)H3_ * H_ * 2);
  _Float16* Wh1h = (_Float16*)alloc((size_t)H3_ * H_ * 2);
  _Float16* Wah  = (_Float16*)alloc((size_t)H_ * H_ * 2);
  _Float16* Uah  = (_Float16*)alloc((size_t)H_ * H_ * 2);
  _Float16* Wi2h = (_Float16*)alloc((size_t)H3_ * 2 * H_ * 2);
  _Float16* Wh2h = (_Float16*)alloc((size_t)H3_ * H_ * 2);
  _Float16* ihh  = (_Float16*)alloc((size_t)B_ * S_ * H_ * 2);
  _Float16* refh = (_Float16*)alloc((size_t)B_ * L_ * H_ * 2);
  _Float16* h1h  = (_Float16*)alloc((size_t)B_ * H_ * 2);
  _Float16* h2h  = (_Float16*)alloc((size_t)B_ * H_ * 2);
  _Float16* cat2 = (_Float16*)alloc((size_t)B_ * 2 * H_ * 2);
  float* instProj = (float*)alloc((size_t)B_ * S_ * H_ * 4);
  float* gi1  = (float*)alloc((size_t)B_ * H3_ * 4);
  float* gh1  = (float*)alloc((size_t)B_ * H3_ * 4);
  float* gi2  = (float*)alloc((size_t)B_ * H3_ * 4);
  float* gh2  = (float*)alloc((size_t)B_ * H3_ * 4);
  float* uahf = (float*)alloc((size_t)B_ * H_ * 4);
  float* h1f  = (float*)alloc((size_t)B_ * H_ * 4);
  float* h2f  = (float*)alloc((size_t)B_ * H_ * 4);

  auto cvt = [&](const float* s, _Float16* d, int n) {
    cvt_f32_to_f16<<<(n + 255) / 256, 256, 0, stream>>>(s, d, n);
  };
  cvt(Wi1, Wi1h, H3_ * H_);
  cvt(Wh1, Wh1h, H3_ * H_);
  cvt(Wa,  Wah,  H_ * H_);
  cvt(Ua,  Uah,  H_ * H_);
  cvt(Wi2, Wi2h, H3_ * 2 * H_);
  cvt(Wh2, Wh2h, H3_ * H_);
  cvt(ih,  ihh,  B_ * S_ * H_);

  embed_gather<<<B_ * L_, H_, 0, stream>>>(instance, solution, Wr, br, refh);
  zero_states<<<(B_ * H_) / 256, 256, 0, stream>>>(h1f, h2f, h1h, h2h);

  // inst_proj = instance_hidden @ Wa^T : (16384,512) x (512,512)
  wmma_gemm_f16<<<dim3(H_ / 256, (B_ * S_) / 16), 256, 0, stream>>>(
      ihh, H_, Wah, nullptr, instProj, H_, H_);

  const dim3 gGate(H3_ / 256, B_ / 16);   // N=1536 GEMMs
  const dim3 gSq(H_ / 256, B_ / 16);      // N=512 GEMMs

  for (int l = 0; l < L_ - 1; ++l) {
    // GRU1 gates: x1 = ref_emb[:, l]
    wmma_gemm_f16<<<gGate, 256, 0, stream>>>(
        refh + (size_t)l * H_, L_ * H_, Wi1h, bi1, gi1, H3_, H_);
    wmma_gemm_f16<<<gGate, 256, 0, stream>>>(
        h1h, H_, Wh1h, bh1, gh1, H3_, H_);
    gru_update<<<(B_ * H_) / 256, 256, 0, stream>>>(gi1, gh1, h1f, h1h);

    // attention query projection + scores/softmax/context
    wmma_gemm_f16<<<gSq, 256, 0, stream>>>(
        h1h, H_, Uah, nullptr, uahf, H_, H_);
    attn_step<<<B_, H_, 0, stream>>>(instProj, uahf, vv, ih, refh, cat2, l);

    // GRU2 gates: input = [x2ref | context] (K = 2H)
    wmma_gemm_f16<<<gGate, 256, 0, stream>>>(
        cat2, 2 * H_, Wi2h, bi2, gi2, H3_, 2 * H_);
    wmma_gemm_f16<<<gGate, 256, 0, stream>>>(
        h2h, H_, Wh2h, bh2, gh2, H3_, H_);
    gru_update<<<(B_ * H_) / 256, 256, 0, stream>>>(gi2, gh2, h2f, h2h);
  }

  final_head<<<B_, 128, 0, stream>>>(h2f, W1, b1, W2, b2, eps, out);
}
